// PointGenerator_49907519980142
// MI455X (gfx1250) — matrix operand order repaired and checked by reading, not compiled, not run on hardware
//
#include <hip/hip_runtime.h>
#include <hip/hip_bf16.h>

#define BB 4
#define PP 512
#define CC 384
#define UPF 4
#define NCTX (BB*PP)        // 2048
#define NTGT (BB*PP*UPF)    // 8192
#define FINF 3.0e38f

typedef __attribute__((ext_vector_type(16))) __bf16 v16bf;
typedef __attribute__((ext_vector_type(8)))  float  v8f;
typedef __attribute__((ext_vector_type(4)))  int    v4i_;

#if __has_builtin(__builtin_amdgcn_global_load_async_to_lds_b128) && \
    __has_builtin(__builtin_amdgcn_s_wait_asynccnt)
#define HAVE_ASYNC_LDS 1
#define AS1 __attribute__((address_space(1)))
#define AS3 __attribute__((address_space(3)))
#endif

// ---- fragment k map used here, contiguous k = 16*hi + e (wave32, 16-bit A 16x32 / B 32x16 / f32 C 16x16) ----
__device__ __forceinline__ int a_k_of(int e, int hi) {
  return hi * 16 + e;
}

// =================== elementwise / prep kernels ===================
__global__ void k_cvt_bf(const float* __restrict__ src, __bf16* __restrict__ dst, int n) {
  int t = blockIdx.x * blockDim.x + threadIdx.x;
  if (t < n) dst[t] = (__bf16)src[t];
}
__global__ void k_zero_bf(__bf16* __restrict__ dst, int n) {
  int t = blockIdx.x * blockDim.x + threadIdx.x;
  if (t < n) dst[t] = (__bf16)0.0f;
}
// transposed weight prep: src f32 [K,N] row-major -> dst bf16 [N][dstK] at column offset kOff
__global__ void k_wcopy_t(__bf16* __restrict__ dst, const float* __restrict__ src,
                          int K, int N, int dstK, int kOff) {
  int t = blockIdx.x * blockDim.x + threadIdx.x;
  if (t >= K * N) return;
  int k = t / N, n = t - k * N;
  dst[(size_t)n * dstK + kOff + k] = (__bf16)src[t];
}
__global__ void k_bias_pad(float* __restrict__ dst, const float* __restrict__ src, int n, int np) {
  int t = blockIdx.x * blockDim.x + threadIdx.x;
  if (t < np) dst[t] = (t < n) ? src[t] : 0.0f;
}
__global__ void k_tgt_tok(const float* __restrict__ ctx_tokens, __bf16* __restrict__ dst) {
  int t = blockIdx.x * blockDim.x + threadIdx.x;
  if (t >= NTGT * CC) return;
  int row = t / CC, c = t - row * CC;
  int b = row / (PP * UPF);
  int w = row - b * (PP * UPF);
  int p = w / UPF;
  dst[t] = (__bf16)ctx_tokens[((size_t)(b * PP + p)) * CC + c];
}
__global__ void k_sqnorm(const __bf16* __restrict__ x, float* __restrict__ sq, int N, int D) {
  int r = blockIdx.x * blockDim.x + threadIdx.x;
  if (r >= N) return;
  float s = 0.0f;
  for (int c = 0; c < D; c++) { float v = (float)x[(size_t)r * D + c]; s += v * v; }
  sq[r] = s;
}
__global__ void k_tgt_xyz0(const float* __restrict__ pred_xyz, const float* __restrict__ noise,
                           float* __restrict__ txyz) {
  int t = blockIdx.x * blockDim.x + threadIdx.x;
  if (t >= NTGT * 3) return;
  int row = t / 3, c = t - row * 3;
  int b = row / (PP * UPF);
  int w = row - b * (PP * UPF);
  int p = w / UPF;
  txyz[t] = pred_xyz[((size_t)(b * PP + p)) * 3 + c] + 0.02f * noise[t];
}
__global__ void k_fold_in(const float* __restrict__ txyz, const __bf16* __restrict__ feat,
                          __bf16* __restrict__ dst) {
  int t = blockIdx.x * blockDim.x + threadIdx.x;
  if (t >= NTGT * 112) return;
  int row = t / 112, c = t - row * 112;
  float v = 0.0f;
  if (c < 3) v = txyz[row * 3 + c];
  else if (c < 99) v = (float)feat[(size_t)row * 96 + (c - 3)];
  dst[t] = (__bf16)v;
}
__global__ void k_ref_in(const float* __restrict__ txyz, const __bf16* __restrict__ feat,
                         __bf16* __restrict__ dst) {
  int t = blockIdx.x * blockDim.x + threadIdx.x;
  if (t >= NTGT * 112) return;
  int row = t / 112, c = t - row * 112;
  float v = 0.0f;
  if (c < 96) v = (float)feat[(size_t)row * 96 + c];
  else if (c < 99) v = txyz[row * 3 + (c - 96)];
  dst[t] = (__bf16)v;
}
__global__ void k_res_add3(float* __restrict__ txyz, const float* __restrict__ h16) {
  int t = blockIdx.x * blockDim.x + threadIdx.x;
  if (t >= NTGT * 3) return;
  int row = t / 3, c = t - row * 3;
  txyz[t] += h16[(size_t)row * 16 + c];
}
__global__ void k_ctx_out(const float* __restrict__ ctx_xyz, const float* __restrict__ t2,
                          float* __restrict__ out) {
  int t = blockIdx.x * blockDim.x + threadIdx.x;
  if (t >= NCTX * 3) return;
  int row = t / 3, c = t - row * 3;
  out[t] = ctx_xyz[t] + 0.05f * t2[(size_t)row * 16 + c];
}
__global__ void k_tgt_out(const float* __restrict__ txyz, const __bf16* __restrict__ rout,
                          float* __restrict__ out) {
  int t = blockIdx.x * blockDim.x + threadIdx.x;
  if (t >= NTGT * 3) return;
  int row = t / 3, c = t - row * 3;
  out[(size_t)NCTX * 3 + t] = txyz[t] + (float)rout[(size_t)row * 16 + c];
}

// =================== brute-force kNN on xyz (D=3) ===================
__global__ void k_knn3(const float* __restrict__ xyz, int N, int* __restrict__ idx) {
  int i = blockIdx.x * blockDim.x + threadIdx.x;
  if (i >= N) return;
  float xi = xyz[3 * i], yi = xyz[3 * i + 1], zi = xyz[3 * i + 2];
  float bd[16]; int bi[16];
  for (int q = 0; q < 16; q++) { bd[q] = FINF; bi[q] = 0; }
  for (int j = 0; j < N; j++) {
    if (j == i) continue;
    float dx = xyz[3 * j] - xi, dy = xyz[3 * j + 1] - yi, dz = xyz[3 * j + 2] - zi;
    float d = dx * dx + dy * dy + dz * dz;
    if (d < bd[15]) {
      int q = 15;
      while (q > 0 && bd[q - 1] > d) { bd[q] = bd[q - 1]; bi[q] = bi[q - 1]; q--; }
      bd[q] = d; bi[q] = j;
    }
  }
  for (int q = 0; q < 16; q++) idx[(size_t)i * 16 + q] = bi[q];
}

// =================== fused WMMA gram + top-k kNN ===================
// one wave per 16-row block; the 16-row j-panel is async-staged into a double-buffered
// LDS tile (GLOBAL_LOAD_ASYNC_TO_LDS_B128) while WMMAs run on the previous panel.
template<int D, int KNN>
__global__ __launch_bounds__(32) void k_knn_wmma(const __bf16* __restrict__ X,
                                                 const float* __restrict__ sq,
                                                 int N, int* __restrict__ idx_out) {
  __shared__ float dtile[16][17];
  const int lane = threadIdx.x;
  const int mloc = lane & 15, hi = lane >> 4;
  const int i0 = blockIdx.x * 16;
  const __bf16* arow = X + (size_t)(i0 + mloc) * D;

  float bd[KNN]; int bi[KNN];
  for (int q = 0; q < KNN; q++) { bd[q] = FINF; bi[q] = 0; }

#ifdef HAVE_ASYNC_LDS
  __shared__ __bf16 bpanel[2][16 * D];
  constexpr int CHUNKS = (16 * D * 2) / 16;      // 16B chunks per panel
  constexpr int BATCH = CHUNKS / 32;             // async instructions per stage
  auto stage = [&](int j0, int buf) {
    AS1 char* gsrc = (AS1 char*)(X + (size_t)j0 * D);
    AS3 char* ldst = (AS3 char*)&bpanel[buf][0];
    #pragma unroll
    for (int c = 0; c < BATCH; c++) {
      int t = c * 32 + lane;
      __builtin_amdgcn_global_load_async_to_lds_b128(
          (AS1 v4i_*)(gsrc + t * 16), (AS3 v4i_*)(ldst + t * 16), 0, 0);
    }
  };
  stage(0, 0);
#endif

  for (int j0 = 0; j0 < N; j0 += 16) {
#ifdef HAVE_ASYNC_LDS
    const int buf = (j0 >> 4) & 1;
    if (j0 + 16 < N) {
      stage(j0 + 16, buf ^ 1);
      __builtin_amdgcn_s_wait_asynccnt(BATCH);   // previous panel complete
    } else {
      __builtin_amdgcn_s_wait_asynccnt(0);
    }
    __syncthreads();
#endif
    v8f acc = {};
    for (int kk = 0; kk < D; kk += 32) {
      v16bf a, b;
#ifdef HAVE_ASYNC_LDS
      const __bf16* brow = &bpanel[buf][mloc * D + kk + hi * 16];
#else
      const __bf16* brow = X + (size_t)(j0 + mloc) * D + kk + hi * 16;
#endif
      #pragma unroll
      for (int e = 0; e < 16; e++) {
        a[e] = arow[kk + a_k_of(e, hi)];
        b[e] = brow[e];
      }
      acc = __builtin_amdgcn_wmma_f32_16x16x32_bf16(false, a, false, b, (short)0, acc, false, false);
    }
    #pragma unroll
    for (int v = 0; v < 8; v++) {
      int m = v + hi * 8;
      float d = sq[i0 + m] + sq[j0 + mloc] - 2.0f * acc[v];
      if (i0 + m == j0 + mloc) d = FINF;
      dtile[m][mloc] = d;
    }
    __syncthreads();
    if (lane < 16) {
      #pragma unroll
      for (int n = 0; n < 16; n++) {
        float d = dtile[lane][n];
        if (d < bd[KNN - 1]) {
          int q = KNN - 1;
          while (q > 0 && bd[q - 1] > d) { bd[q] = bd[q - 1]; bi[q] = bi[q - 1]; q--; }
          bd[q] = d; bi[q] = j0 + n;
        }
      }
    }
    __syncthreads();
  }
  if (lane < 16)
    for (int q = 0; q < KNN; q++) idx_out[(size_t)(i0 + lane) * KNN + q] = bi[q];
}

// =================== fused EdgeConv: mlp2 over edges + segment max ===================
// one wave per tile of 16 edges; A-fragment built once per K-step and reused across
// ALL output N-tiles (multi-accumulator). Weights are pre-transposed: Wt[n][k].
template<int DIN, int H, int HOUT, int KNN>
__global__ __launch_bounds__(32) void k_edgeconv(const __bf16* __restrict__ X,
                                                 const int* __restrict__ idx,
                                                 const __bf16* __restrict__ W1t,
                                                 const float* __restrict__ B1,
                                                 const __bf16* __restrict__ W2t,
                                                 const float* __restrict__ B2,
                                                 __bf16* __restrict__ out) {
  constexpr int NT1 = H / 16, NT2 = HOUT / 16;
  constexpr int K1 = 2 * DIN;
  __shared__ __bf16 h1[16][H];
  __shared__ float  otile[16][17];
  const int lane = threadIdx.x;
  const int mloc = lane & 15, hi = lane >> 4;
  const long e0 = (long)blockIdx.x * 16;
  const int iA = (int)((e0 + mloc) / KNN);
  const int jA = idx[e0 + mloc];
  const __bf16* xi = X + (size_t)iA * DIN;
  const __bf16* xj = X + (size_t)jA * DIN;
  __builtin_prefetch(xi, 0, 1);
  __builtin_prefetch(xj, 0, 1);

  // GEMM1: [16, 2*DIN] x W1t, ReLU+bias -> LDS (bf16)
  {
    v8f acc[NT1] = {};
    for (int kk = 0; kk < K1; kk += 32) {
      v16bf a;
      #pragma unroll
      for (int e = 0; e < 16; e++) {
        int k = kk + a_k_of(e, hi);
        float f = (k < DIN) ? (float)xi[k] : ((float)xj[k - DIN] - (float)xi[k - DIN]);
        a[e] = (__bf16)f;
      }
      #pragma unroll
      for (int nt = 0; nt < NT1; nt++) {
        v16bf b;
        const __bf16* brow = W1t + (size_t)(nt * 16 + mloc) * K1 + kk + hi * 16;
        #pragma unroll
        for (int e = 0; e < 16; e++) b[e] = brow[e];
        acc[nt] = __builtin_amdgcn_wmma_f32_16x16x32_bf16(false, a, false, b, (short)0, acc[nt], false, false);
      }
    }
    #pragma unroll
    for (int nt = 0; nt < NT1; nt++)
      #pragma unroll
      for (int v = 0; v < 8; v++) {
        int m = v + hi * 8;
        float t = acc[nt][v] + B1[nt * 16 + mloc];
        h1[m][nt * 16 + mloc] = (__bf16)fmaxf(t, 0.0f);
      }
  }
  __syncthreads();

  // GEMM2: [16, H] x W2t, bias, then column-wise max over KNN-edge groups
  v8f acc2[NT2] = {};
  for (int kk = 0; kk < H; kk += 32) {
    v16bf a;
    #pragma unroll
    for (int e = 0; e < 16; e++) a[e] = h1[mloc][kk + a_k_of(e, hi)];
    #pragma unroll
    for (int nt = 0; nt < NT2; nt++) {
      v16bf b;
      const __bf16* brow = W2t + (size_t)(nt * 16 + mloc) * H + kk + hi * 16;
      #pragma unroll
      for (int e = 0; e < 16; e++) b[e] = brow[e];
      acc2[nt] = __builtin_amdgcn_wmma_f32_16x16x32_bf16(false, a, false, b, (short)0, acc2[nt], false, false);
    }
  }
  #pragma unroll
  for (int nt = 0; nt < NT2; nt++) {
    #pragma unroll
    for (int v = 0; v < 8; v++)
      otile[v + hi * 8][mloc] = acc2[nt][v] + B2[nt * 16 + mloc];
    __syncthreads();
    if (lane < 16) {
      constexpr int PPT = 16 / KNN;  // points per 16-edge tile
      for (int p = 0; p < PPT; p++) {
        float mx = -FINF;
        for (int r = 0; r < KNN; r++) mx = fmaxf(mx, otile[p * KNN + r][lane]);
        long pt = e0 / KNN + p;
        out[(size_t)pt * HOUT + nt * 16 + lane] = (__bf16)mx;
      }
    }
    __syncthreads();
  }
}

// =================== generic WMMA GEMM: out = act(A Wt^T + b) ===================
// NT output tiles per wave reuse one A fragment; Wt is [Nn][K] (transposed).
template<int NT, bool RELU, bool BF16OUT>
__global__ __launch_bounds__(32) void k_gemm(const __bf16* __restrict__ A,
                                             const __bf16* __restrict__ Wt,
                                             const float* __restrict__ bias,
                                             void* __restrict__ outp,
                                             int M, int K, int Nn) {
  const int groups_n = Nn / (16 * NT);
  const int tm = blockIdx.x / groups_n, tg = blockIdx.x - tm * groups_n;
  const int lane = threadIdx.x;
  const int mloc = lane & 15, hi = lane >> 4;
  const __bf16* arow = A + (size_t)(tm * 16 + mloc) * K;
  v8f acc[NT] = {};
  for (int kk = 0; kk < K; kk += 32) {
    v16bf a;
    #pragma unroll
    for (int e = 0; e < 16; e++) a[e] = arow[kk + a_k_of(e, hi)];
    #pragma unroll
    for (int nt = 0; nt < NT; nt++) {
      v16bf b;
      const __bf16* brow = Wt + (size_t)((tg * NT + nt) * 16 + mloc) * K + kk + hi * 16;
      #pragma unroll
      for (int e = 0; e < 16; e++) b[e] = brow[e];
      acc[nt] = __builtin_amdgcn_wmma_f32_16x16x32_bf16(false, a, false, b, (short)0, acc[nt], false, false);
    }
  }
  #pragma unroll
  for (int nt = 0; nt < NT; nt++)
    #pragma unroll
    for (int v = 0; v < 8; v++) {
      int m = tm * 16 + v + hi * 8;
      int n = (tg * NT + nt) * 16 + mloc;
      float t = acc[nt][v] + bias[n];
      if (RELU) t = fmaxf(t, 0.0f);
      if (BF16OUT) ((__bf16*)outp)[(size_t)m * Nn + n] = (__bf16)t;
      else         ((float*)outp)[(size_t)m * Nn + n] = t;
    }
}

// =================== host ===================
#define CDIV(a, b) (((a) + (b) - 1) / (b))

extern "C" void kernel_launch(void* const* d_in, const int* in_sizes, int n_in,
                              void* d_out, int out_size, void* d_ws, size_t ws_size,
                              hipStream_t stream) {
  (void)in_sizes; (void)n_in; (void)out_size; (void)ws_size;
  const float* ctx_xyz   = (const float*)d_in[0];
  const float* ctx_tok   = (const float*)d_in[1];
  const float* pred_xyz  = (const float*)d_in[2];
  const float* noise     = (const float*)d_in[4];
  const float* c1w1 = (const float*)d_in[5];  const float* c1b1 = (const float*)d_in[6];
  const float* c1w2 = (const float*)d_in[7];  const float* c1b2 = (const float*)d_in[8];
  const float* c2w1 = (const float*)d_in[9];  const float* c2b1 = (const float*)d_in[10];
  const float* c2w2 = (const float*)d_in[11]; const float* c2b2 = (const float*)d_in[12];
  const float* cdw1 = (const float*)d_in[13]; const float* cdb1 = (const float*)d_in[14];
  const float* cdw2 = (const float*)d_in[15]; const float* cdb2 = (const float*)d_in[16];
  const float* fw1  = (const float*)d_in[17]; const float* fb1  = (const float*)d_in[18];
  const float* fw2  = (const float*)d_in[19]; const float* fb2  = (const float*)d_in[20];
  const float* fw3  = (const float*)d_in[21]; const float* fb3  = (const float*)d_in[22];
  const float* rw1  = (const float*)d_in[23]; const float* rb1  = (const float*)d_in[24];
  const float* rw2  = (const float*)d_in[25]; const float* rb2  = (const float*)d_in[26];
  float* out = (float*)d_out;

  char* cur = (char*)d_ws;
  auto alloc = [&](size_t bytes) -> void* {
    void* p = (void*)cur; cur += (bytes + 255) & ~(size_t)255; return p;
  };
  __bf16* ctx_bf  = (__bf16*)alloc((size_t)NCTX * CC * 2);
  __bf16* tgt_bf  = (__bf16*)alloc((size_t)NTGT * CC * 2);
  float*  sqbuf   = (float*) alloc((size_t)NTGT * 4);
  int*    idx1t   = (int*)   alloc((size_t)NTGT * 16 * 4);
  int*    idx1c   = (int*)   alloc((size_t)NCTX * 16 * 4);
  int*    idx2t   = (int*)   alloc((size_t)NTGT * 8 * 4);
  int*    idx2c   = (int*)   alloc((size_t)NCTX * 8 * 4);
  int*    idxr    = (int*)   alloc((size_t)NTGT * 16 * 4);
  __bf16* h1t     = (__bf16*)alloc((size_t)NTGT * 192 * 2);
  __bf16* h1c     = (__bf16*)alloc((size_t)NCTX * 192 * 2);
  __bf16* featt   = (__bf16*)alloc((size_t)NTGT * 96 * 2);
  __bf16* featc   = (__bf16*)alloc((size_t)NCTX * 96 * 2);
  __bf16* t1      = (__bf16*)alloc((size_t)NCTX * 96 * 2);
  float*  t2      = (float*) alloc((size_t)NCTX * 16 * 4);
  float*  txyz    = (float*) alloc((size_t)NTGT * 3 * 4);
  __bf16* fin     = (__bf16*)alloc((size_t)NTGT * 112 * 2);
  __bf16* fh1     = (__bf16*)alloc((size_t)NTGT * 256 * 2);
  __bf16* fh2     = (__bf16*)alloc((size_t)NTGT * 256 * 2);
  float*  fh3     = (float*) alloc((size_t)NTGT * 16 * 4);
  __bf16* rin     = (__bf16*)alloc((size_t)NTGT * 112 * 2);
  __bf16* rout    = (__bf16*)alloc((size_t)NTGT * 16 * 2);
  // transposed (and padded) bf16 weights: Wt[N][K]
  __bf16* wc11t = (__bf16*)alloc(192 * 768 * 2);
  __bf16* wc12t = (__bf16*)alloc(192 * 192 * 2);
  __bf16* wc21t = (__bf16*)alloc(96 * 384 * 2);
  __bf16* wc22t = (__bf16*)alloc(96 * 96 * 2);
  __bf16* wcd1t = (__bf16*)alloc(96 * 96 * 2);
  __bf16* wcd2t = (__bf16*)alloc(16 * 96 * 2);
  __bf16* wf1t  = (__bf16*)alloc(256 * 112 * 2);
  __bf16* wf2t  = (__bf16*)alloc(256 * 256 * 2);
  __bf16* wf3t  = (__bf16*)alloc(16 * 256 * 2);
  __bf16* wr1t  = (__bf16*)alloc(384 * 224 * 2);
  __bf16* wr2t  = (__bf16*)alloc(16 * 384 * 2);
  float* bcd2 = (float*)alloc(16 * 4);
  float* bf3p = (float*)alloc(16 * 4);
  float* br2p = (float*)alloc(16 * 4);

  const int EW = 256;
  #define EL(n) CDIV((n), EW), EW, 0, stream

  // ---- weight prep (transpose + pad + bf16) ----
  k_wcopy_t<<<EL(768 * 192)>>>(wc11t, c1w1, 768, 192, 768, 0);
  k_wcopy_t<<<EL(192 * 192)>>>(wc12t, c1w2, 192, 192, 192, 0);
  k_wcopy_t<<<EL(384 * 96)>>>(wc21t, c2w1, 384, 96, 384, 0);
  k_wcopy_t<<<EL(96 * 96)>>>(wc22t, c2w2, 96, 96, 96, 0);
  k_wcopy_t<<<EL(96 * 96)>>>(wcd1t, cdw1, 96, 96, 96, 0);
  k_zero_bf<<<EL(16 * 96)>>>(wcd2t, 16 * 96);
  k_wcopy_t<<<EL(96 * 3)>>>(wcd2t, cdw2, 96, 3, 96, 0);
  k_zero_bf<<<EL(256 * 112)>>>(wf1t, 256 * 112);
  k_wcopy_t<<<EL(99 * 256)>>>(wf1t, fw1, 99, 256, 112, 0);
  k_wcopy_t<<<EL(256 * 256)>>>(wf2t, fw2, 256, 256, 256, 0);
  k_zero_bf<<<EL(16 * 256)>>>(wf3t, 16 * 256);
  k_wcopy_t<<<EL(256 * 3)>>>(wf3t, fw3, 256, 3, 256, 0);
  k_zero_bf<<<EL(384 * 224)>>>(wr1t, 384 * 224);
  k_wcopy_t<<<EL(99 * 384)>>>(wr1t, rw1, 99, 384, 224, 0);
  k_wcopy_t<<<EL(99 * 384)>>>(wr1t, rw1 + (size_t)99 * 384, 99, 384, 224, 112);
  k_zero_bf<<<EL(16 * 384)>>>(wr2t, 16 * 384);
  k_wcopy_t<<<EL(384 * 3)>>>(wr2t, rw2, 384, 3, 384, 0);
  k_bias_pad<<<EL(16)>>>(bcd2, cdb2, 3, 16);
  k_bias_pad<<<EL(16)>>>(bf3p, fb3, 3, 16);
  k_bias_pad<<<EL(16)>>>(br2p, rb2, 3, 16);

  // ---- token prep ----
  k_cvt_bf<<<EL(NCTX * CC)>>>(ctx_tok, ctx_bf, NCTX * CC);
  k_tgt_tok<<<EL(NTGT * CC)>>>(ctx_tok, tgt_bf);

  // ---- shared EdgeConv: layer 1 (knn on 384-d tokens, mlp 768->192->192, max over 16) ----
  k_sqnorm<<<EL(NTGT)>>>(tgt_bf, sqbuf, NTGT, CC);
  k_knn_wmma<384, 16><<<NTGT / 16, 32, 0, stream>>>(tgt_bf, sqbuf, NTGT, idx1t);
  k_sqnorm<<<EL(NCTX)>>>(ctx_bf, sqbuf, NCTX, CC);
  k_knn_wmma<384, 16><<<NCTX / 16, 32, 0, stream>>>(ctx_bf, sqbuf, NCTX, idx1c);
  k_edgeconv<384, 192, 192, 16><<<NTGT * 16 / 16, 32, 0, stream>>>(tgt_bf, idx1t, wc11t, c1b1, wc12t, c1b2, h1t);
  k_edgeconv<384, 192, 192, 16><<<NCTX * 16 / 16, 32, 0, stream>>>(ctx_bf, idx1c, wc11t, c1b1, wc12t, c1b2, h1c);

  // ---- shared EdgeConv: layer 2 (knn on 192-d h, mlp 384->96->96, max over 8) ----
  k_sqnorm<<<EL(NTGT)>>>(h1t, sqbuf, NTGT, 192);
  k_knn_wmma<192, 8><<<NTGT / 16, 32, 0, stream>>>(h1t, sqbuf, NTGT, idx2t);
  k_sqnorm<<<EL(NCTX)>>>(h1c, sqbuf, NCTX, 192);
  k_knn_wmma<192, 8><<<NCTX / 16, 32, 0, stream>>>(h1c, sqbuf, NCTX, idx2c);
  k_edgeconv<192, 96, 96, 8><<<NTGT * 8 / 16, 32, 0, stream>>>(h1t, idx2t, wc21t, c2b1, wc22t, c2b2, featt);
  k_edgeconv<192, 96, 96, 8><<<NCTX * 8 / 16, 32, 0, stream>>>(h1c, idx2c, wc21t, c2b1, wc22t, c2b2, featc);

  // ---- context head: ctx_xyz + 0.05 * mlp2(cdef)(ctx_feat) ----
  k_gemm<2, true, true><<<(NCTX / 16) * (96 / 32), 32, 0, stream>>>(featc, wcd1t, cdb1, (void*)t1, NCTX, 96, 96);
  k_gemm<1, false, false><<<(NCTX / 16) * 1, 32, 0, stream>>>(t1, wcd2t, bcd2, (void*)t2, NCTX, 96, 16);
  k_ctx_out<<<EL(NCTX * 3)>>>(ctx_xyz, t2, out);

  // ---- folding MLP on target ----
  k_tgt_xyz0<<<EL(NTGT * 3)>>>(pred_xyz, noise, txyz);
  k_fold_in<<<EL(NTGT * 112)>>>(txyz, featt, fin);
  k_gemm<4, true, true><<<(NTGT / 16) * (256 / 64), 32, 0, stream>>>(fin, wf1t, fb1, (void*)fh1, NTGT, 112, 256);
  k_gemm<4, true, true><<<(NTGT / 16) * (256 / 64), 32, 0, stream>>>(fh1, wf2t, fb2, (void*)fh2, NTGT, 256, 256);
  k_gemm<1, false, false><<<(NTGT / 16) * 1, 32, 0, stream>>>(fh2, wf3t, bf3p, (void*)fh3, NTGT, 256, 16);
  k_res_add3<<<EL(NTGT * 3)>>>(txyz, fh3);

  // ---- GNN refiner ----
  k_knn3<<<EL(NTGT)>>>(txyz, NTGT, idxr);
  k_ref_in<<<EL(NTGT * 112)>>>(txyz, featt, rin);
  k_edgeconv<112, 384, 16, 16><<<NTGT * 16 / 16, 32, 0, stream>>>(rin, idxr, wr1t, rb1, wr2t, br2p, rout);
  k_tgt_out<<<EL(NTGT * 3)>>>(txyz, rout, out);
}
